// WassersteinDistance_80041010528774
// MI455X (gfx1250) — matrix operand order repaired and checked by reading, not compiled, not run on hardware
//
#include <hip/hip_runtime.h>
#include <hip/hip_bf16.h>
#include <math.h>

#define BATCH 16
#define LS 1024
#define LT 1024
#define DD 512
#define NEG_INF_F (-1e14f)
#define LAMBDA_W 0.1f

typedef __attribute__((ext_vector_type(16))) __bf16 v16bf;
typedef __attribute__((ext_vector_type(8)))  float  v8f;

union FragBf { v16bf v; uint4 q[2]; };

// ---------------------------------------------------------------------------
// Prep: W (513x513 f32) -> Wb (512x512 bf16), only the [k<512, l<512] block.
// ---------------------------------------------------------------------------
__global__ void wd_conv_w(const float* __restrict__ W, __bf16* __restrict__ Wb) {
    int i = blockIdx.x * blockDim.x + threadIdx.x;
    if (i < DD * DD) {
        int k = i >> 9, l = i & 511;
        Wb[i] = (__bf16)W[k * 513 + l];
    }
}

// ---------------------------------------------------------------------------
// Prep target: f32 -> bf16, and per-row squared norm t2.
// One block (128 threads) per row of 512.
// ---------------------------------------------------------------------------
__global__ void wd_prep_tgt(const float* __restrict__ t, __bf16* __restrict__ tb,
                            float* __restrict__ t2) {
    const int row = blockIdx.x;                  // b*LT + ti
    const int tid = threadIdx.x;                 // 128
    const float* p = t + (size_t)row * DD;
    __bf16* q = tb + (size_t)row * DD;
    float4 x = ((const float4*)p)[tid];
    q[tid * 4 + 0] = (__bf16)x.x;
    q[tid * 4 + 1] = (__bf16)x.y;
    q[tid * 4 + 2] = (__bf16)x.z;
    q[tid * 4 + 3] = (__bf16)x.w;
    float acc = x.x * x.x + x.y * x.y + x.z * x.z + x.w * x.w;
    #pragma unroll
    for (int off = 1; off < 32; off <<= 1) acc += __shfl_xor(acc, off, 32);
    __shared__ float ws[4];
    if ((tid & 31) == 0) ws[tid >> 5] = acc;
    __syncthreads();
    if (tid == 0) t2[row] = ws[0] + ws[1] + ws[2] + ws[3];
}

// ---------------------------------------------------------------------------
// Prep source: f32 -> bf16, per-row squared norm s2, and
// vbias[row] = sum_l W[512,l]*s[row,l] + W[512,512]  (k=512 column of v).
// ---------------------------------------------------------------------------
__global__ void wd_prep_src(const float* __restrict__ s, const float* __restrict__ W,
                            __bf16* __restrict__ sb, float* __restrict__ s2,
                            float* __restrict__ vbias) {
    const int row = blockIdx.x;                  // b*LS + si
    const int tid = threadIdx.x;                 // 128
    const float* p = s + (size_t)row * DD;
    const float* wr = W + (size_t)512 * 513;     // row 512 of W
    __bf16* q = sb + (size_t)row * DD;
    float4 x = ((const float4*)p)[tid];
    float4 w = ((const float4*)wr)[tid];
    q[tid * 4 + 0] = (__bf16)x.x;
    q[tid * 4 + 1] = (__bf16)x.y;
    q[tid * 4 + 2] = (__bf16)x.z;
    q[tid * 4 + 3] = (__bf16)x.w;
    float acc = x.x * x.x + x.y * x.y + x.z * x.z + x.w * x.w;
    float vac = x.x * w.x + x.y * w.y + x.z * w.z + x.w * w.w;
    #pragma unroll
    for (int off = 1; off < 32; off <<= 1) {
        acc += __shfl_xor(acc, off, 32);
        vac += __shfl_xor(vac, off, 32);
    }
    __shared__ float ws[4], wv[4];
    if ((tid & 31) == 0) { ws[tid >> 5] = acc; wv[tid >> 5] = vac; }
    __syncthreads();
    if (tid == 0) {
        s2[row] = ws[0] + ws[1] + ws[2] + ws[3];
        vbias[row] = wv[0] + wv[1] + wv[2] + wv[3] + wr[512];
    }
}

// ---------------------------------------------------------------------------
// Mask sums per batch + zero the per-batch accumulator.
// ---------------------------------------------------------------------------
__global__ void wd_prep_mask(const float* __restrict__ smask, const float* __restrict__ tmask,
                             float* __restrict__ nL1, float* __restrict__ nL2,
                             float* __restrict__ Sacc) {
    const int b = blockIdx.x;
    const int tid = threadIdx.x;                 // 256
    float a = 0.f, c = 0.f;
    for (int i = tid; i < LS; i += 256) {
        a += smask[b * LS + i];
        c += tmask[b * LT + i];
    }
    #pragma unroll
    for (int off = 1; off < 32; off <<= 1) {
        a += __shfl_xor(a, off, 32);
        c += __shfl_xor(c, off, 32);
    }
    __shared__ float ra[8], rc[8];
    if ((tid & 31) == 0) { ra[tid >> 5] = a; rc[tid >> 5] = c; }
    __syncthreads();
    if (tid == 0) {
        float sa = 0.f, sc = 0.f;
        #pragma unroll
        for (int i = 0; i < 8; ++i) { sa += ra[i]; sc += rc[i]; }
        nL2[b] = sa;        // sum(source_mask)
        nL1[b] = sc;        // sum(target_mask)
        Sacc[b] = 0.f;
    }
}

// ---------------------------------------------------------------------------
// v GEMM: v[b,s,k] = sum_{l<512} src[b,s,l] * W[k,l] + W[k,512], bf16 out.
// Block = 256 thr (8 waves). Wave w does a 16(s) x 16(k) tile, K=512 chain.
// grid = (B, LS/16, 512/128)
// ---------------------------------------------------------------------------
__global__ void wd_vgemm(const __bf16* __restrict__ srcb, const __bf16* __restrict__ Wb,
                         const float* __restrict__ W, __bf16* __restrict__ vb) {
    const int b = blockIdx.x;
    const int s0 = blockIdx.y * 16;
    const int wave = threadIdx.x >> 5;
    const int lane = threadIdx.x & 31;
    const int half = lane >> 4;
    const int idx  = lane & 15;
    const int kcol0 = blockIdx.z * 128 + wave * 16;

    const __bf16* sb = srcb + ((size_t)b * LS + s0) * DD;
    v8f c = {};
    #pragma unroll 4
    for (int l0 = 0; l0 < DD; l0 += 32) {
        FragBf a, bw;
        a.q[0] = *(const uint4*)&sb[(size_t)idx * DD + l0 + half * 16];
        a.q[1] = *(const uint4*)&sb[(size_t)idx * DD + l0 + half * 16 + 8];
        const __bf16* wp = &Wb[(size_t)(kcol0 + idx) * DD + l0 + half * 16];
        bw.q[0] = *(const uint4*)wp;
        bw.q[1] = *(const uint4*)(wp + 8);
        c = __builtin_amdgcn_wmma_f32_16x16x32_bf16(false, a.v, false, bw.v,
                                                    (short)0, c, false, false);
    }
    const int kc = kcol0 + idx;
    const float bias = W[(size_t)kc * 513 + 512];   // s_ext bias column
    __bf16* out = vb + (size_t)b * LS * DD;
    #pragma unroll
    for (int r = 0; r < 8; ++r) {
        const int row = s0 + r + half * 8;          // C layout: M = r + 8*half
        out[(size_t)row * DD + kc] = (__bf16)(c[r] + bias);
    }
}

// ---------------------------------------------------------------------------
// Main fused kernel: per block = one (b, 16-row t tile).
// 8 waves; wave w streams s-columns [w*128, w*128+128) in PAIRS of 16-wide
// tiles (4 WMMA chains per k-step, A-fragment reused 4x). Online softmax
// state kept PER LANE (each lane owns one column per tile), merged across the
// 16-lane group once at the end, then across waves in LDS.
// grid = (B, LT/16), block = 256
// ---------------------------------------------------------------------------
__global__ void wd_main(const __bf16* __restrict__ tgtb, const __bf16* __restrict__ srcb,
                        const __bf16* __restrict__ vb, const float* __restrict__ vbias,
                        const float* __restrict__ t2, const float* __restrict__ s2,
                        const float* __restrict__ smask, const float* __restrict__ tmask,
                        float* __restrict__ Sacc) {
    __shared__ __bf16 ldsT[16 * DD];                // 16 KB target tile
    __shared__ float pm[8][16], pz[8][16], pa[8][16];
    __shared__ float t2s[16], tms[16], red[16];

    const int b  = blockIdx.x;
    const int t0 = blockIdx.y * 16;
    const int tid  = threadIdx.x;
    const int wave = tid >> 5;
    const int lane = tid & 31;
    const int half = lane >> 4;
    const int idx  = lane & 15;

    // Stage the contiguous 16x512 bf16 target tile into LDS.
    const __bf16* tg = tgtb + ((size_t)b * LT + t0) * DD;
    for (int i = tid; i < 16 * DD / 8; i += 256)
        ((uint4*)ldsT)[i] = ((const uint4*)tg)[i];
    if (tid < 16) {
        t2s[tid] = t2[b * LT + t0 + tid];
        tms[tid] = tmask[b * LT + t0 + tid];
    }
    __syncthreads();

    // Hoist per-row constants for this lane's row set (rows half*8 + r).
    float t2r[8], tmr[8];
    #pragma unroll
    for (int r = 0; r < 8; ++r) {
        t2r[r] = t2s[half * 8 + r];
        tmr[r] = tms[half * 8 + r];
    }

    const __bf16* sb = srcb + (size_t)b * LS * DD;
    const __bf16* vv = vb  + (size_t)b * LS * DD;

    // Per-lane online softmax state (this lane's 8 rows x its column stream).
    float m[8], Z[8], A[8];
    #pragma unroll
    for (int r = 0; r < 8; ++r) { m[r] = -__builtin_inff(); Z[r] = 0.f; A[r] = 0.f; }

    const int sBase = wave * 128;
    for (int sp = 0; sp < 4; ++sp) {
        const int s0 = sBase + sp * 32;
        v8f cD[2] = {{}, {}}, cT[2] = {{}, {}};
        const __bf16* sr0 = &sb[(size_t)(s0 + idx) * DD];
        const __bf16* sr1 = &sb[(size_t)(s0 + 16 + idx) * DD];
        const __bf16* vr0 = &vv[(size_t)(s0 + idx) * DD];
        const __bf16* vr1 = &vv[(size_t)(s0 + 16 + idx) * DD];
        #pragma unroll 4
        for (int k0 = 0; k0 < DD; k0 += 32) {
            FragBf a, b0, b1, b2, b3;
            // A fragment from LDS: row = idx, K contiguous {k0+half*16..+15}
            a.q[0] = *(const uint4*)&ldsT[idx * DD + k0 + half * 16];
            a.q[1] = *(const uint4*)&ldsT[idx * DD + k0 + half * 16 + 8];
            const int ko = k0 + half * 16;
            b0.q[0] = *(const uint4*)(sr0 + ko); b0.q[1] = *(const uint4*)(sr0 + ko + 8);
            b1.q[0] = *(const uint4*)(vr0 + ko); b1.q[1] = *(const uint4*)(vr0 + ko + 8);
            b2.q[0] = *(const uint4*)(sr1 + ko); b2.q[1] = *(const uint4*)(sr1 + ko + 8);
            b3.q[0] = *(const uint4*)(vr1 + ko); b3.q[1] = *(const uint4*)(vr1 + ko + 8);
            cD[0] = __builtin_amdgcn_wmma_f32_16x16x32_bf16(false, a.v, false, b0.v,
                                                            (short)0, cD[0], false, false);
            cT[0] = __builtin_amdgcn_wmma_f32_16x16x32_bf16(false, a.v, false, b1.v,
                                                            (short)0, cT[0], false, false);
            cD[1] = __builtin_amdgcn_wmma_f32_16x16x32_bf16(false, a.v, false, b2.v,
                                                            (short)0, cD[1], false, false);
            cT[1] = __builtin_amdgcn_wmma_f32_16x16x32_bf16(false, a.v, false, b3.v,
                                                            (short)0, cT[1], false, false);
        }
        if (sp + 1 < 4) {   // global_prefetch_b8 for the next tile pair
            __builtin_prefetch(&sb[(size_t)(s0 + 32 + idx) * DD], 0, 0);
            __builtin_prefetch(&vv[(size_t)(s0 + 32 + idx) * DD], 0, 0);
        }
        #pragma unroll
        for (int t = 0; t < 2; ++t) {
            const int scol = s0 + t * 16 + idx;
            const float s2c = s2[b * LS + scol];
            const float vbc = vbias[b * LS + scol];
            const float smc = smask[b * LS + scol];
            #pragma unroll
            for (int r = 0; r < 8; ++r) {
                float tr = cT[t][r] + vbc;
                if (tmr[r] * smc == 0.f) tr = NEG_INF_F;
                float cst = t2r[r] + s2c - 2.f * cD[t][r];
                cst = sqrtf(fmaxf(cst, 0.f));
                // per-lane online softmax update (no cross-lane traffic here)
                const float nm = fmaxf(m[r], tr);
                const float e0 = __expf(m[r] - nm);
                const float p  = __expf(tr - nm);
                Z[r] = Z[r] * e0 + p;
                A[r] = A[r] * e0 + p * cst;
                m[r] = nm;
            }
        }
    }

    // Single cross-lane merge of (m,Z,A) within the 16-lane group.
    #pragma unroll
    for (int off = 1; off < 16; off <<= 1) {
        #pragma unroll
        for (int r = 0; r < 8; ++r) {
            const float om = __shfl_xor(m[r], off, 16);
            const float oz = __shfl_xor(Z[r], off, 16);
            const float oa = __shfl_xor(A[r], off, 16);
            const float nm = fmaxf(m[r], om);
            const float e0 = __expf(m[r] - nm);
            const float e1 = __expf(om - nm);
            Z[r] = Z[r] * e0 + oz * e1;
            A[r] = A[r] * e0 + oa * e1;
            m[r] = nm;
        }
    }

    // Per-wave partials -> LDS (lanes 0 and 16 carry rows 0-7 / 8-15).
    if (idx == 0) {
        #pragma unroll
        for (int r = 0; r < 8; ++r) {
            pm[wave][half * 8 + r] = m[r];
            pz[wave][half * 8 + r] = Z[r];
            pa[wave][half * 8 + r] = A[r];
        }
    }
    __syncthreads();
    if (tid < 16) {
        float mm = -__builtin_inff(), zz = 0.f, aa = 0.f;
        #pragma unroll
        for (int w = 0; w < 8; ++w) {
            const float wm = pm[w][tid], wz = pz[w][tid], wa = pa[w][tid];
            const float nm = fmaxf(mm, wm);
            const float e1 = __expf(mm - nm), e2 = __expf(wm - nm);
            zz = zz * e1 + wz * e2;
            aa = aa * e1 + wa * e2;
            mm = nm;
        }
        red[tid] = aa / zz;     // sum over s of softmax * cost, for this t row
    }
    __syncthreads();
    if (tid == 0) {
        float sum = 0.f;
        #pragma unroll
        for (int i = 0; i < 16; ++i) sum += red[i];
        atomicAdd(&Sacc[b], sum);
    }
}

// ---------------------------------------------------------------------------
// Final: mean over the broadcast [B,B] of S[b1]/(nL1[b1]*nL2[b2]) * lambda.
// = lambda/B^2 * (sum_b S/nL1) * (sum_b 1/nL2)
// ---------------------------------------------------------------------------
__global__ void wd_final(const float* __restrict__ Sacc, const float* __restrict__ nL1,
                         const float* __restrict__ nL2, float* __restrict__ out) {
    if (threadIdx.x == 0 && blockIdx.x == 0) {
        float s1 = 0.f, s2 = 0.f;
        #pragma unroll
        for (int bq = 0; bq < BATCH; ++bq) {
            s1 += Sacc[bq] / nL1[bq];
            s2 += 1.0f / nL2[bq];
        }
        out[0] = LAMBDA_W * s1 * s2 / (float)(BATCH * BATCH);
    }
}

// ---------------------------------------------------------------------------
extern "C" void kernel_launch(void* const* d_in, const int* in_sizes, int n_in,
                              void* d_out, int out_size, void* d_ws, size_t ws_size,
                              hipStream_t stream) {
    const float* source = (const float*)d_in[0];   // [16,1024,512]
    const float* target = (const float*)d_in[1];   // [16,1024,512]
    const float* smask  = (const float*)d_in[2];   // [16,1024]
    const float* tmask  = (const float*)d_in[3];   // [16,1024]
    const float* W      = (const float*)d_in[4];   // [513,513]
    float* out = (float*)d_out;

    char* ws = (char*)d_ws;
    size_t off = 0;
    auto alloc = [&](size_t bytes) -> char* {
        char* p = ws + off;
        off = (off + bytes + 255) & ~(size_t)255;
        return p;
    };
    const size_t NBD = (size_t)BATCH * LS * DD;         // 8M elements
    __bf16* srcb  = (__bf16*)alloc(NBD * 2);
    __bf16* tgtb  = (__bf16*)alloc(NBD * 2);
    __bf16* vb    = (__bf16*)alloc(NBD * 2);
    __bf16* Wb    = (__bf16*)alloc((size_t)DD * DD * 2);
    float*  t2    = (float*)alloc((size_t)BATCH * LT * 4);
    float*  s2    = (float*)alloc((size_t)BATCH * LS * 4);
    float*  vbias = (float*)alloc((size_t)BATCH * LS * 4);
    float*  nL1   = (float*)alloc(BATCH * 4);
    float*  nL2   = (float*)alloc(BATCH * 4);
    float*  Sacc  = (float*)alloc(BATCH * 4);
    (void)ws_size; (void)in_sizes; (void)n_in; (void)out_size;

    wd_conv_w<<<(DD * DD + 255) / 256, 256, 0, stream>>>(W, Wb);
    wd_prep_tgt<<<BATCH * LT, 128, 0, stream>>>(target, tgtb, t2);
    wd_prep_src<<<BATCH * LS, 128, 0, stream>>>(source, W, srcb, s2, vbias);
    wd_prep_mask<<<BATCH, 256, 0, stream>>>(smask, tmask, nL1, nL2, Sacc);
    wd_vgemm<<<dim3(BATCH, LS / 16, DD / 128), 256, 0, stream>>>(srcb, Wb, W, vb);
    wd_main<<<dim3(BATCH, LT / 16), 256, 0, stream>>>(tgtb, srcb, vb, vbias,
                                                      t2, s2, smask, tmask, Sacc);
    wd_final<<<1, 32, 0, stream>>>(Sacc, nL1, nL2, out);
}
